// MoEDense_1271310319711
// MI455X (gfx1250) — matrix operand order repaired and checked by reading, not compiled, not run on hardware
//
#include <hip/hip_runtime.h>

#define B_  64
#define HW_ 4096
#define C_  256
#define E_  8

typedef __attribute__((ext_vector_type(16))) _Float16     v16h;
typedef __attribute__((ext_vector_type(8)))  float        v8f;
typedef __attribute__((ext_vector_type(4)))  unsigned int v4u;

union H16 { v4u u[2]; v16h h; };

// ---------------- Kernel 1: partial global-average pooling -----------------
// grid (16, 64) blocks of 256 threads; block (s, b) sums 256 spatial positions
// for all 256 channels. Fully coalesced (256 threads span one 1KB channel row).
__global__ __launch_bounds__(256)
void pool_partial(const float* __restrict__ x, float* __restrict__ pp) {
  const int s = blockIdx.x, b = blockIdx.y, c = threadIdx.x;
  const float* xp = x + ((size_t)b * HW_ + (size_t)s * 256) * C_ + c;
  float acc = 0.f;
  #pragma unroll 4
  for (int p = 0; p < 256; ++p) acc += xp[(size_t)p * C_];
  pp[((size_t)b * 16 + s) * C_ + c] = acc;
}

// ---------------- Kernel 2: finish pool, gate, argmax, LB loss -------------
__global__ __launch_bounds__(256)
void gate_kernel(const float* __restrict__ pp,
                 const float* __restrict__ gW,   // [C_IN][E]
                 const float* __restrict__ gb,   // [E]
                 float* __restrict__ pooled,     // ws [B][C]
                 int*   __restrict__ eidx,       // ws [B]
                 float* __restrict__ lb_out) {
  const int t = threadIdx.x;
  // Phase A: reduce 16 partials per (b,c), scale by 1/(H*W). Coalesced.
  for (int idx = t; idx < B_ * C_; idx += 256) {
    const int b = idx >> 8, c = idx & 255;
    float s = 0.f;
    for (int k = 0; k < 16; ++k) s += pp[((size_t)b * 16 + k) * C_ + c];
    pooled[idx] = s * (1.0f / 4096.0f);
  }
  __syncthreads();
  __shared__ int sh_idx[B_];
  // Phase B: one thread per batch computes 8 gate logits + argmax.
  if (t < B_) {
    float logit[E_];
    #pragma unroll
    for (int e = 0; e < E_; ++e) logit[e] = gb[e];
    const float* pb = pooled + t * C_;
    for (int c = 0; c < C_; ++c) {
      const float pv = pb[c];
      #pragma unroll
      for (int e = 0; e < E_; ++e) logit[e] += pv * gW[c * E_ + e];
    }
    int bi = 0; float bv = logit[0];
    #pragma unroll
    for (int e = 1; e < E_; ++e) if (logit[e] > bv) { bv = logit[e]; bi = e; }
    eidx[t]   = bi;   // softmax is monotonic -> argmax(logits) == argmax(probs)
    sh_idx[t] = bi;
  }
  __syncthreads();
  // Phase C: KL(usage || uniform)
  if (t == 0) {
    float cnt[E_];
    #pragma unroll
    for (int e = 0; e < E_; ++e) cnt[e] = 0.f;
    for (int b = 0; b < B_; ++b) cnt[sh_idx[b]] += 1.f;
    float usage[E_], tot = 0.f;
    #pragma unroll
    for (int e = 0; e < E_; ++e) { usage[e] = cnt[e] / (float)B_ + 1e-6f; tot += usage[e]; }
    float lb = 0.f;
    const float lu = logf(1.0f / (float)E_);
    #pragma unroll
    for (int e = 0; e < E_; ++e) {
      const float u = usage[e] / tot;
      lb += u * (logf(u) - lu);
    }
    *lb_out = lb;
  }
}

// ---------------- Kernel 3: expert GEMM via WMMA ---------------------------
// grid (64 Mblocks, 64 batches), 256 threads = 8 waves.
// Block computes out[b, m0:m0+64, 0:256]; wave (mt = w%4, nh = w/4) owns the
// 16x128 strip [m0+16*mt, cols nh*128 .. nh*128+127] as 8 f32 accumulators.
__global__ __launch_bounds__(256)
void moe_gemm(const float* __restrict__ x,
              const float* __restrict__ W,     // [E][C][C]
              const float* __restrict__ bias,  // [E][C]
              const int*   __restrict__ eidx,
              float* __restrict__ out) {
  const int m0   = blockIdx.x * 64;
  const int b    = blockIdx.y;
  const int tid  = threadIdx.x;
  const int wave = tid >> 5;
  const int lane = tid & 31;
  const int ln16 = lane & 15;
  const int hi   = lane >> 4;       // 0: lanes 0-15, 1: lanes 16-31
  const int mt   = wave & 3;
  const int nh   = wave >> 2;

  const int e = eidx[b];
  const float* Wp = W + (size_t)e * C_ * C_;

  // K-contiguous f16 tiles; row stride 40 halves = 80B (16B aligned,
  // 20-bank stride -> conflict-free ds_load_b128 across 16 rows).
  __shared__ _Float16 Xs[64][40];    // [m][k]
  __shared__ _Float16 Ws[256][40];   // [n][k]  (transposed weight slab)

  v8f zero = {};
  v8f acc[8];
  #pragma unroll
  for (int i = 0; i < 8; ++i) acc[i] = zero;

  const float* xrow = x + ((size_t)b * HW_ + m0) * C_;

  for (int k0 = 0; k0 < C_; k0 += 32) {
    // Stage X chunk (64x32): thread t -> kk=t&31, rows t>>5 + 8*i. Coalesced.
    {
      const int kk = tid & 31;
      const int mb = tid >> 5;
      #pragma unroll
      for (int i = 0; i < 8; ++i) {
        const int m = mb + i * 8;
        Xs[m][kk] = (_Float16)xrow[(size_t)m * C_ + (k0 + kk)];
      }
    }
    // Stage W chunk transposed (Ws[n][kk] = W[e][k0+kk][n]); reads coalesced
    // over n = tid for each kk.
    {
      const int n = tid;
      #pragma unroll
      for (int kk = 0; kk < 32; ++kk)
        Ws[n][kk] = (_Float16)Wp[(size_t)(k0 + kk) * C_ + n];
    }
    __syncthreads();

    // A fragment: 16x32 f16 per ISA layout:
    //   lanes 0-15: VGPR0-3 = K 0..7, VGPR4-7 = K 16..23 (row = M = lane)
    //   lanes16-31: VGPR0-3 = K 8..15, VGPR4-7 = K 24..31
    H16 A;
    const int arow = mt * 16 + ln16;
    const int akb  = hi * 8;
    A.u[0] = *(const v4u*)&Xs[arow][akb];
    A.u[1] = *(const v4u*)&Xs[arow][akb + 16];

    #pragma unroll
    for (int nt = 0; nt < 8; ++nt) {
      // B fragment: 32x16 f16: lanes 0-15 hold K 0..7,16..23 (col N = lane),
      // lanes 16-31 hold K 8..15,24..31 — same K order as A -> two b128 loads.
      const int n   = nh * 128 + nt * 16 + ln16;
      const int bkb = hi * 8;
      H16 Bf;
      Bf.u[0] = *(const v4u*)&Ws[n][bkb];
      Bf.u[1] = *(const v4u*)&Ws[n][bkb + 16];
      acc[nt] = __builtin_amdgcn_wmma_f32_16x16x32_f16(
          false, A.h, false, Bf.h, (short)0, acc[nt], false, false);
    }
    __syncthreads();
  }

  // Epilogue: C/D layout: lanes 0-15 -> M = r, N = lane; lanes 16-31 -> M = 8+r.
  const float* bp  = bias + e * C_;
  float* orow = out + ((size_t)b * HW_ + m0) * C_;
  #pragma unroll
  for (int nt = 0; nt < 8; ++nt) {
    const int col = nh * 128 + nt * 16 + ln16;
    const float bv = bp[col];
    #pragma unroll
    for (int r = 0; r < 8; ++r) {
      const int m = mt * 16 + r + hi * 8;
      orow[(size_t)m * C_ + col] = acc[nt][r] + bv;
    }
  }
}

extern "C" void kernel_launch(void* const* d_in, const int* in_sizes, int n_in,
                              void* d_out, int out_size, void* d_ws, size_t ws_size,
                              hipStream_t stream) {
  const float* x  = (const float*)d_in[0];   // [64,64,64,256]
  const float* W  = (const float*)d_in[1];   // [8,256,256]
  const float* bb = (const float*)d_in[2];   // [8,256]
  const float* gW = (const float*)d_in[3];   // [256,8]
  const float* gb = (const float*)d_in[4];   // [8]
  float* out = (float*)d_out;

  char* ws = (char*)d_ws;
  float* pp     = (float*)ws;                       // 64*16*256*4 = 1 MB
  float* pooled = (float*)(ws + (1 << 20));         // 64*256*4   = 64 KB
  int*   eidx   = (int*)  (ws + (1 << 20) + (1 << 16));
  float* lb_out = out + (size_t)B_ * HW_ * C_;      // d_out[67108864]

  pool_partial<<<dim3(16, 64), 256, 0, stream>>>(x, pp);
  gate_kernel <<<dim3(1),      256, 0, stream>>>(pp, gW, gb, pooled, eidx, lb_out);
  moe_gemm    <<<dim3(64, 64), 256, 0, stream>>>(x, W, bb, eidx, out);
}
